// minLSTM_51539608288
// MI455X (gfx1250) — matrix operand order repaired and checked by reading, not compiled, not run on hardware
//
#include <hip/hip_runtime.h>
#include <hip/hip_bf16.h>
#include <math.h>

// Problem shape (fixed by reference): B=4, S=8192, D=1024
#define BB 4
#define SS 8192
#define DD 1024
#define NN 3072              // 3*D
#define MM 32768             // B*S
#define CHUNK 256            // scan chunk length
#define NCHUNK (SS / CHUNK)  // 32 chunks per channel
#define NCHAN (BB * DD)      // 4096 independent scan channels
#define KB 32                // K per bf16 WMMA
#define NKB (DD / KB)        // 32 K-blocks

// GEMM blocking: 512 threads = 16 waves = 8 M-tiles x 2 d-tiles per block
#define BLK_MT 8
#define BLK_DT 2
#define STAGE_A (BLK_MT * 2 * 1024)            // 16 KB (hi+lo)
#define STAGE_B (BLK_DT * 3 * 2 * 1024)        // 12 KB (hi+lo, 3 gates)
#define STAGE_BYTES (STAGE_A + STAGE_B)        // 28 KB

typedef __attribute__((ext_vector_type(16))) __bf16 v16bf;
typedef __attribute__((ext_vector_type(8)))  float  v8f;

#define NEG_INF (-1.0e30f)

#if defined(__gfx1250__) && __has_builtin(__builtin_amdgcn_tensor_load_to_lds) && \
    __has_builtin(__builtin_amdgcn_s_wait_tensorcnt)
#define USE_TDM 1
typedef __attribute__((ext_vector_type(4))) unsigned int u32x4;
typedef __attribute__((ext_vector_type(8))) int i32x8;
typedef __attribute__((ext_vector_type(4))) int i32x4;
#endif

__device__ __forceinline__ float softplus_f(float x) {
    return fmaxf(x, 0.0f) + log1pf(expf(-fabsf(x)));
}

__device__ __forceinline__ float logaddexp_f(float a, float b) {
    float m = fmaxf(a, b);
    float d = fminf(a, b) - m;
    return m + log1pf(expf(d));
}

__device__ __forceinline__ void decomp_bf16(float v, __bf16* hi, __bf16* lo) {
    __bf16 h = (__bf16)v;
    *hi = h;
    *lo = (__bf16)(v - (float)h);
}

#define WMMA_BF16(A, B, C) \
    __builtin_amdgcn_wmma_f32_16x16x32_bf16(false, (A), false, (B), (short)0, (C), false, false)

// ---------------------------------------------------------------------------
// One-time hi/lo bf16 decomposition into WMMA fragment layout.
//   X*: [mt(2048)][kb(32)][lane(32)][j(16)]
//   W*: [g(3)*64 + dt][kb(32)][lane(32)][j(16)]
// ---------------------------------------------------------------------------
__global__ __launch_bounds__(256) void pack_x(
    const float* __restrict__ x, __bf16* __restrict__ Xhi, __bf16* __restrict__ Xlo)
{
    int t = blockIdx.x * blockDim.x + threadIdx.x;
    int lane = t & 31;
    int kb   = (t >> 5) & 31;
    int mt   = t >> 10;

    int m = mt * 16 + (lane & 15);
    int h = lane >> 4;
    size_t obase = ((size_t)(mt * 32 + kb) * 32 + lane) * 16;
    const float* __restrict__ xr = x + (size_t)m * DD + kb * 32;

    #pragma unroll
    for (int j = 0; j < 16; ++j) {
        int kk = (j < 8) ? (h * 8 + j) : (16 + h * 8 + (j - 8));
        __bf16 hi, lo;
        decomp_bf16(xr[kk], &hi, &lo);
        Xhi[obase + j] = hi;
        Xlo[obase + j] = lo;
    }
}

__global__ __launch_bounds__(256) void pack_w(
    const float* __restrict__ W, __bf16* __restrict__ Whi, __bf16* __restrict__ Wlo)
{
    int t = blockIdx.x * blockDim.x + threadIdx.x;
    int lane = t & 31;
    int kb   = (t >> 5) & 31;
    int dt   = (t >> 10) & 63;
    int g    = t >> 16;

    int n = g * DD + dt * 16 + (lane & 15);
    int k0 = kb * 32;
    int h = lane >> 4;
    size_t obase = ((size_t)((g * 64 + dt) * 32 + kb) * 32 + lane) * 16;

    #pragma unroll
    for (int j = 0; j < 16; ++j) {
        __bf16 hi, lo;
        int kk = (j < 8) ? (h * 8 + j) : (16 + h * 8 + (j - 8));
        decomp_bf16(W[(size_t)(k0 + kk) * NN + n], &hi, &lo);
        Whi[obase + j] = hi;
        Wlo[obase + j] = lo;
    }
}

#ifdef USE_TDM
// Issue one TDM tensor load: rows of (tile0q * 8) bytes, tile1 x tile2 rows,
// row strides s0q / s1q (all in 8-byte units). Writes LDS contiguously.
__device__ __forceinline__ void tdm_load(
    unsigned lds_addr, unsigned long long ga,
    unsigned tile0q, unsigned tile1, unsigned tile2,
    unsigned long long s0q, unsigned long long s1q)
{
    unsigned td0 = 0x400000u;   // generous tensor_dim0 (no OOB clipping)
    unsigned td1 = 0x100000u;   // generous tensor_dim1

    u32x4 g0;
    g0[0] = 1u;                                          // count=1, user mode
    g0[1] = lds_addr;                                    // lds_addr
    g0[2] = (unsigned)ga;                                // global_addr[31:0]
    g0[3] = ((unsigned)(ga >> 32) & 0x01FFFFFFu) | 0x80000000u;  // type=2

    i32x8 g1;
    g1[0] = (int)(3u << 16);                             // data_size = 8 bytes
    g1[1] = (int)((td0 & 0xFFFFu) << 16);                // tensor_dim0[15:0]
    g1[2] = (int)(((td0 >> 16) & 0xFFFFu) | ((td1 & 0xFFFFu) << 16));
    g1[3] = (int)(((td1 >> 16) & 0xFFFFu) | (tile0q << 16));   // tile_dim0
    g1[4] = (int)(tile1 | (tile2 << 16));                // tile_dim1, tile_dim2
    g1[5] = (int)(unsigned)s0q;                          // tensor_dim0_stride[31:0]
    g1[6] = (int)((unsigned)((s0q >> 32) & 0xFFFFu) | (((unsigned)s1q & 0xFFFFu) << 16));
    g1[7] = (int)(unsigned)(s1q >> 16);                  // tensor_dim1_stride[47:16]

    i32x4 g2;
    g2[0] = 16;                                          // tensor_dim2 (generous)
    g2[1] = 0; g2[2] = 0; g2[3] = 0;
    i32x4 g3 = (i32x4)0;
    i32x8 g4 = (i32x8)0;   // extra arg on clang-23 6-arg form (unused features)

    __builtin_amdgcn_tensor_load_to_lds(g0, g1, g2, g3, g4, 0);
}
#endif

// ---------------------------------------------------------------------------
// GEMM via bf16x3 fp32 emulation with block-level LDS staging.
// Stage layout (per buffer): A: [hl(2)][m(8)][1KB]  B: [hl(2)][gate(3)][dtl(2)][1KB]
// ---------------------------------------------------------------------------
__global__ __launch_bounds__(512) void minlstm_gemm_gates(
    const __bf16* __restrict__ Xhi, const __bf16* __restrict__ Xlo,
    const __bf16* __restrict__ Whi, const __bf16* __restrict__ Wlo,
    float* __restrict__ lf, float* __restrict__ lv)
{
    __shared__ __align__(16) char smem[2][STAGE_BYTES];

    const int tid  = threadIdx.x;
    const int lane = tid & 31;
    const int w    = tid >> 5;          // 16 waves
    const int mi   = w >> 1;            // 0..7  (M-tile in block)
    const int di   = w & 1;             // 0..1  (d-tile in block)

    const int mt_base = (blockIdx.x >> 5) * BLK_MT;   // 256 M-blocks
    const int dt_base = (blockIdx.x & 31) * BLK_DT;   // 32 d-blocks

    v8f accH = {}, accF = {}, accI = {};

    // ---- stage loader -----------------------------------------------------
    auto load_stage = [&](int buf, int kb) {
#ifdef USE_TDM
        unsigned ldsA = (unsigned)(uintptr_t)&smem[buf][0];
        unsigned ldsB = ldsA + STAGE_A;
        if (w == 0) {
            // A: 8 rows (m) of 1KB, row stride 32KB; hi then lo
            unsigned long long gaH = (unsigned long long)(uintptr_t)Xhi +
                ((unsigned long long)(mt_base * 32 + kb)) * 1024ull;
            unsigned long long gaL = (unsigned long long)(uintptr_t)Xlo +
                ((unsigned long long)(mt_base * 32 + kb)) * 1024ull;
            tdm_load(ldsA,        gaH, 128, 8, 0, 4096, 0);
            tdm_load(ldsA + 8192, gaL, 128, 8, 0, 4096, 0);
        } else if (w == 1) {
            // B: 3 gates (Z, stride 2MB) x 2 d-tiles (Y, stride 32KB) x 1KB rows
            unsigned long long gbH = (unsigned long long)(uintptr_t)Whi +
                ((unsigned long long)(dt_base * 32 + kb)) * 1024ull;
            unsigned long long gbL = (unsigned long long)(uintptr_t)Wlo +
                ((unsigned long long)(dt_base * 32 + kb)) * 1024ull;
            tdm_load(ldsB,        gbH, 128, 2, 3, 4096, 262144);
            tdm_load(ldsB + 6144, gbL, 128, 2, 3, 4096, 262144);
        }
#else
        char* dst = &smem[buf][0];
        for (int li = tid; li < STAGE_BYTES / 16; li += 512) {
            const __bf16* src;
            if (li < 1024) {                       // A region: [hl][m][64x16B]
                int hl  = li >> 9;
                int m   = (li >> 6) & 7;
                int off = li & 63;
                const __bf16* base = hl ? Xlo : Xhi;
                src = base + (size_t)((mt_base + m) * 32 + kb) * 512 + off * 8;
            } else {                               // B region: [hl][gate][dtl][64x16B]
                int lj  = li - 1024;
                int hl  = lj / 384;
                int r   = lj % 384;
                int g   = r >> 7;
                int dtl = (r >> 6) & 1;
                int off = r & 63;
                const __bf16* base = hl ? Wlo : Whi;
                src = base + (size_t)((g * 64 + dt_base + dtl) * 32 + kb) * 512 + off * 8;
            }
            *(uint4*)(dst + li * 16) = *(const uint4*)src;
        }
#endif
    };

    auto wait_stage = [&]() {
#ifdef USE_TDM
        __builtin_amdgcn_s_wait_tensorcnt(0);
#endif
        __syncthreads();
    };

    // ---- main loop: double-buffered K-blocks ------------------------------
    load_stage(0, 0);
    wait_stage();

    int buf = 0;
    for (int kb = 0; kb < NKB; ++kb) {
        if (kb + 1 < NKB) load_stage(buf ^ 1, kb + 1);

        const char* sA = &smem[buf][0];
        const char* sB = sA + STAGE_A;

        v16bf Ah = *(const v16bf*)(sA +        mi * 1024 + lane * 32);
        v16bf Al = *(const v16bf*)(sA + 8192 + mi * 1024 + lane * 32);

        v16bf BHh = *(const v16bf*)(sB +        0 * 2048 + di * 1024 + lane * 32);
        v16bf BFh = *(const v16bf*)(sB +        1 * 2048 + di * 1024 + lane * 32);
        v16bf BIh = *(const v16bf*)(sB +        2 * 2048 + di * 1024 + lane * 32);
        v16bf BHl = *(const v16bf*)(sB + 6144 + 0 * 2048 + di * 1024 + lane * 32);
        v16bf BFl = *(const v16bf*)(sB + 6144 + 1 * 2048 + di * 1024 + lane * 32);
        v16bf BIl = *(const v16bf*)(sB + 6144 + 2 * 2048 + di * 1024 + lane * 32);

        accH = WMMA_BF16(Ah, BHh, accH);
        accF = WMMA_BF16(Ah, BFh, accF);
        accI = WMMA_BF16(Ah, BIh, accI);
        accH = WMMA_BF16(Ah, BHl, accH);
        accF = WMMA_BF16(Ah, BFl, accF);
        accI = WMMA_BF16(Ah, BIl, accI);
        accH = WMMA_BF16(Al, BHh, accH);
        accF = WMMA_BF16(Al, BFh, accF);
        accI = WMMA_BF16(Al, BIh, accI);

        wait_stage();
        buf ^= 1;
    }

    // ---- fused epilogue ---------------------------------------------------
    const int half = lane >> 4;
    const int rbase = (mt_base + mi) * 16 + half * 8;
    const int ccol  = (dt_base + di) * 16 + (lane & 15);
    #pragma unroll
    for (int r = 0; r < 8; ++r) {
        float h = accH[r];
        float f = accF[r];
        float i = accI[r];

        float diff  = softplus_f(-f) - softplus_f(-i);
        float log_f = -softplus_f(diff);
        float log_i = -softplus_f(-diff);
        float log_g = (h >= 0.0f) ? logf(h + 0.5f) : -softplus_f(-h);

        size_t o = (size_t)(rbase + r) * DD + ccol;
        lf[o] = log_f;
        lv[o] = log_i + log_g;
    }
}

// ---------------------------------------------------------------------------
// Chunked Heinsen scan: h_t = f_t*h_{t-1} + v_t in log space.
// ---------------------------------------------------------------------------
__global__ __launch_bounds__(256) void minlstm_chunk_agg(
    const float* __restrict__ lf, const float* __restrict__ lv,
    float* __restrict__ aggF, float* __restrict__ aggV)
{
    int idx = blockIdx.x * blockDim.x + threadIdx.x;
    int d     = idx & (DD - 1);
    int chunk = (idx >> 10) & (NCHUNK - 1);
    int b     = idx >> 15;

    size_t base = ((size_t)b * SS + (size_t)chunk * CHUNK) * DD + d;

    float F = 0.0f;
    float V = NEG_INF;
    for (int t = 0; t < CHUNK; ++t) {
        float f = lf[base + (size_t)t * DD];
        float v = lv[base + (size_t)t * DD];
        F += f;
        V = logaddexp_f(V + f, v);
    }
    size_t ai = ((size_t)b * DD + d) * NCHUNK + chunk;
    aggF[ai] = F;
    aggV[ai] = V;
}

__global__ __launch_bounds__(256) void minlstm_carry_scan(
    const float* __restrict__ aggF, const float* __restrict__ aggV,
    float* __restrict__ carry)
{
    int ch = blockIdx.x * blockDim.x + threadIdx.x;
    if (ch >= NCHAN) return;
    size_t base = (size_t)ch * NCHUNK;
    float c = NEG_INF;
    for (int k = 0; k < NCHUNK; ++k) {
        carry[base + k] = c;
        c = logaddexp_f(aggF[base + k] + c, aggV[base + k]);
    }
}

__global__ __launch_bounds__(256) void minlstm_apply(
    const float* __restrict__ lf, const float* __restrict__ lv,
    const float* __restrict__ carry, float* __restrict__ out)
{
    int idx = blockIdx.x * blockDim.x + threadIdx.x;
    int d     = idx & (DD - 1);
    int chunk = (idx >> 10) & (NCHUNK - 1);
    int b     = idx >> 15;

    size_t base = ((size_t)b * SS + (size_t)chunk * CHUNK) * DD + d;
    float lh = carry[((size_t)b * DD + d) * NCHUNK + chunk];

    for (int t = 0; t < CHUNK; ++t) {
        size_t o = base + (size_t)t * DD;
        float f = lf[o];
        float v = lv[o];
        lh = logaddexp_f(lh + f, v);
        out[o] = expf(lh);
    }
}

// ---------------------------------------------------------------------------
extern "C" void kernel_launch(void* const* d_in, const int* in_sizes, int n_in,
                              void* d_out, int out_size, void* d_ws, size_t ws_size,
                              hipStream_t stream) {
    const float* x = (const float*)d_in[0];   // [B,S,D] fp32
    const float* W = (const float*)d_in[1];   // [D,3D] fp32
    float* out = (float*)d_out;               // [B,S,D] fp32

    float* lf    = (float*)d_ws;                       // 128 MB
    float* lv    = lf + (size_t)MM * DD;               // 128 MB
    float* aggF  = lv + (size_t)MM * DD;               // 512 KB
    float* aggV  = aggF + (size_t)NCHAN * NCHUNK;      // 512 KB
    float* carry = aggV + (size_t)NCHAN * NCHUNK;      // 512 KB
    __bf16* Xhi  = (__bf16*)(carry + (size_t)NCHAN * NCHUNK);  // 64 MB
    __bf16* Xlo  = Xhi + (size_t)MM * DD;              // 64 MB
    __bf16* Whi  = Xlo + (size_t)MM * DD;              // 6 MB
    __bf16* Wlo  = Whi + (size_t)DD * NN;              // 6 MB

    pack_x<<<dim3((2048 * 32 * 32) / 256), 256, 0, stream>>>(x, Xhi, Xlo);
    pack_w<<<dim3((3 * 64 * 32 * 32) / 256), 256, 0, stream>>>(W, Whi, Wlo);

    // 256 M-blocks x 32 d-blocks, 16 waves/block (8 M-tiles x 2 d-tiles)
    minlstm_gemm_gates<<<dim3(256 * 32), 512, 0, stream>>>(Xhi, Xlo, Whi, Wlo, lf, lv);

    minlstm_chunk_agg<<<dim3((NCHAN * NCHUNK) / 256), 256, 0, stream>>>(lf, lv, aggF, aggV);
    minlstm_carry_scan<<<dim3((NCHAN + 255) / 256), 256, 0, stream>>>(aggF, aggV, carry);
    minlstm_apply<<<dim3((NCHAN * NCHUNK) / 256), 256, 0, stream>>>(lf, lv, carry, out);
}
